// StageNet_24472723652651
// MI455X (gfx1250) — compile-verified, matrix operand order checked by reading, not run on hardware
//
#include <hip/hip_runtime.h>

// ---------------- constants ----------------
#define BB 256
#define TT 256
#define INF_ 256
#define HH 1024
#define LL 8
#define CSZ 10
#define OUTD 64
#define GG 4112         // 4*H + 2*L
#define GP 4160         // GG padded to multiple of 64 (N padding so GEMM loads need no guard)
#define KC 1312         // 257 + 1025 padded to multiple of 32
#define KH 10240        // H * CS
#define HS 170          // H // 6

typedef __attribute__((ext_vector_type(16))) __bf16 v16bf;
typedef __attribute__((ext_vector_type(8)))  float  v8f;

__device__ __forceinline__ __bf16 f2bf(float f) {
    union { float f; unsigned u; } x; x.f = f;
    unsigned r = (x.u + 0x7FFFu + ((x.u >> 16) & 1u)) >> 16;
    unsigned short s = (unsigned short)r;
    __bf16 o; __builtin_memcpy(&o, &s, 2); return o;
}
__device__ __forceinline__ float bf2f(__bf16 b) {
    unsigned short s; __builtin_memcpy(&s, &b, 2);
    union { unsigned u; float f; } x; x.u = ((unsigned)s) << 16; return x.f;
}
__device__ __forceinline__ float sigmoidf(float x) { return 1.0f / (1.0f + __expf(-x)); }

union FragU { v16bf v; int4 q[2]; };

// ---------------- generic bf16 WMMA GEMM ----------------
// C(MxN,f32) = A(MxK,bf16,row-major, lda=LD) * Bt(NxK,bf16,row-major, ldb=LD) + bias
// Bt must have >= gridDim.x*64 valid (possibly zero-padded) rows so the K-loop is
// branch-free; stores are guarded by N.
// MODE: 0 = bias only, 1 = bias + sigmoid
// Block = 128 threads (4 waves). Wave w: rows [(by*4+w)*16, +16), cols bx*64..+64
template<int LD, int K, int MODE>
__global__ void wmma_gemm_bf16(const __bf16* __restrict__ A,
                               const __bf16* __restrict__ Bt,
                               float* __restrict__ C, int ldc,
                               const float* __restrict__ bias,
                               int M, int N) {
    const int lane = threadIdx.x & 31;
    const int wave = threadIdx.x >> 5;
    const int m0 = (blockIdx.y * 4 + wave) * 16;
    if (m0 >= M) return;
    const int n0base = blockIdx.x * 64;
    const int half = lane >> 4;
    const int idx  = lane & 15;

    const __bf16* ap = A  + (size_t)(m0 + idx) * LD + half * 8;
    const __bf16* bp = Bt + (size_t)(n0base + idx) * LD + half * 8;

    v8f zz = {0.f,0.f,0.f,0.f,0.f,0.f,0.f,0.f};
    v8f acc0 = zz, acc1 = zz, acc2 = zz, acc3 = zz;

    for (int kk = 0; kk < K; kk += 32) {
        FragU a, b0, b1, b2, b3;
        a.q[0]  = *(const int4*)(ap);
        a.q[1]  = *(const int4*)(ap + 16);
        b0.q[0] = *(const int4*)(bp);
        b0.q[1] = *(const int4*)(bp + 16);
        b1.q[0] = *(const int4*)(bp + 16 * LD);
        b1.q[1] = *(const int4*)(bp + 16 * LD + 16);
        b2.q[0] = *(const int4*)(bp + 32 * LD);
        b2.q[1] = *(const int4*)(bp + 32 * LD + 16);
        b3.q[0] = *(const int4*)(bp + 48 * LD);
        b3.q[1] = *(const int4*)(bp + 48 * LD + 16);
        acc0 = __builtin_amdgcn_wmma_f32_16x16x32_bf16(false, a.v, false, b0.v, (short)0, acc0, false, false);
        acc1 = __builtin_amdgcn_wmma_f32_16x16x32_bf16(false, a.v, false, b1.v, (short)0, acc1, false, false);
        acc2 = __builtin_amdgcn_wmma_f32_16x16x32_bf16(false, a.v, false, b2.v, (short)0, acc2, false, false);
        acc3 = __builtin_amdgcn_wmma_f32_16x16x32_bf16(false, a.v, false, b3.v, (short)0, acc3, false, false);
        ap += 32;
        bp += 32;
    }

    v8f accs[4] = {acc0, acc1, acc2, acc3};
#pragma unroll
    for (int i = 0; i < 4; ++i) {
        int n0 = n0base + i * 16;
        if (n0 >= N) continue;               // wave-uniform guard, store-only
        int col = n0 + idx;
        float bv = bias ? bias[col] : 0.f;
#pragma unroll
        for (int r = 0; r < 8; ++r) {
            int row = m0 + half * 8 + r;
            float v = accs[i][r] + bv;
            if (MODE == 1) v = sigmoidf(v);
            C[(size_t)row * ldc + col] = v;
        }
    }
}

// ---------------- one-time prep ----------------
__global__ void k_zero_f(float* p, size_t n) {
    size_t i = (size_t)blockIdx.x * blockDim.x + threadIdx.x;
    if (i < n) p[i] = 0.f;
}
__global__ void k_cvt_bf16(const float* __restrict__ s, __bf16* __restrict__ d, size_t n) {
    size_t i = (size_t)blockIdx.x * blockDim.x + threadIdx.x;
    if (i < n) d[i] = f2bf(s[i]);
}
// Wcomb_t: GP rows x KC cols. Row n<GG: k<257 -> kernel_w[k][n]; 257..1281 -> rec_w[k-257][n]; else 0.
// Rows n>=GG are zero padding so GEMM loads need no N guard.
__global__ void k_build_wcomb(const float* __restrict__ kw, const float* __restrict__ rw,
                              __bf16* __restrict__ wt) {
    size_t i = (size_t)blockIdx.x * blockDim.x + threadIdx.x;
    if (i >= (size_t)GP * KC) return;
    int n = (int)(i / KC), k = (int)(i % KC);
    float v = 0.f;
    if (n < GG) {
        if (k < 257)       v = kw[(size_t)k * GG + n];
        else if (k < 1282) v = rw[(size_t)(k - 257) * GG + n];
    }
    wt[i] = f2bf(v);
}
__global__ void k_bias_g(const float* __restrict__ kb, const float* __restrict__ rb,
                         float* __restrict__ bg) {
    int i = blockIdx.x * blockDim.x + threadIdx.x;
    if (i < GG) bg[i] = kb[i] + rb[i];
}
// Wflat_t[o][k*H + c] = conv_w[o][c][k]
__global__ void k_build_wflat(const float* __restrict__ cw, __bf16* __restrict__ wt) {
    size_t i = (size_t)blockIdx.x * blockDim.x + threadIdx.x;
    if (i >= (size_t)HH * KH) return;
    int o = (int)(i / KH), j = (int)(i % KH);
    int k = j >> 10, c = j & 1023;
    wt[i] = f2bf(cw[((size_t)o * HH + c) * CSZ + k]);
}
// owt[o][c] = out_w[c][o]
__global__ void k_build_owt(const float* __restrict__ ow, __bf16* __restrict__ wt) {
    int i = blockIdx.x * blockDim.x + threadIdx.x;
    if (i >= OUTD * HH) return;
    int o = i >> 10, c = i & 1023;
    wt[i] = f2bf(ow[(size_t)c * OUTD + o]);
}

// ---------------- per-step kernels ----------------
// A[b][0:256]=x_t, [256]=time, [257:1281]=h, [1281]=time, rest 0
__global__ void k_buildA(const __bf16* __restrict__ xb, const float* __restrict__ tm,
                         const __bf16* __restrict__ hbf, __bf16* __restrict__ A, int t) {
    int i = blockIdx.x * blockDim.x + threadIdx.x;
    if (i >= BB * KC) return;
    int b = i / KC, k = i % KC;
    __bf16 v;
    if (k < 256)        v = xb[((size_t)b * TT + t) * INF_ + k];
    else if (k == 256)  v = f2bf(tm[(size_t)b * TT + t]);
    else if (k < 1281)  v = hbf[(size_t)b * HH + (k - 257)];
    else if (k == 1281) v = f2bf(tm[(size_t)b * TT + t]);
    else                v = f2bf(0.f);
    A[i] = v;
}

// gates: per batch row. Updates c, h(bf16), ring_h, ring_dis, distance output.
__global__ void k_gates(const float* __restrict__ xo, float* __restrict__ c,
                        __bf16* __restrict__ hbf, __bf16* __restrict__ ring_h,
                        float* __restrict__ ring_dis, float* __restrict__ dist_out, int t) {
    int b = blockIdx.x;
    const float* xob = xo + (size_t)b * GG;
    __shared__ float fm[LL], im[LL];
    if (threadIdx.x == 0) {
        float m0 = -1e30f, m1 = -1e30f;
        for (int l = 0; l < LL; ++l) { m0 = fmaxf(m0, xob[l]); m1 = fmaxf(m1, xob[LL + l]); }
        float e0[LL], e1[LL], s0 = 0.f, s1 = 0.f;
        for (int l = 0; l < LL; ++l) {
            e0[l] = __expf(xob[l] - m0);       s0 += e0[l];
            e1[l] = __expf(xob[LL + l] - m1);  s1 += e1[l];
        }
        float cs = 0.f, fsum = 0.f;
        for (int l = 0; l < LL; ++l) { cs += e0[l] / s0; fm[l] = cs; fsum += cs; }
        float suf = 0.f;
        for (int l = LL - 1; l >= 0; --l) { suf += e1[l] / s1; im[l] = suf; }
        float cur_dis = 1.0f - fsum / (float)LL;
        ring_dis[(t % CSZ) * BB + b] = cur_dis;
        dist_out[(size_t)t * BB + b] = cur_dis;
    }
    __syncthreads();
    int slot = t % CSZ;
    for (int j = threadIdx.x; j < HH; j += blockDim.x) {
        int l = j >> 7;
        float fg = sigmoidf(xob[2 * LL + j]);
        float ig = sigmoidf(xob[2 * LL + HH + j]);
        float og = sigmoidf(xob[2 * LL + 2 * HH + j]);
        float ci = tanhf   (xob[2 * LL + 3 * HH + j]);
        float cl = c[(size_t)b * HH + j];
        float F = fm[l], I = im[l], ov = F * I;
        float cn = ov * (fg * cl + ig * ci) + (F - ov) * cl + (I - ov) * ci;
        float hn = og * tanhf(cn);
        c[(size_t)b * HH + j] = cn;
        __bf16 hb = f2bf(hn);
        hbf[(size_t)b * HH + j] = hb;
        ring_h[((size_t)slot * BB + b) * HH + j] = hb;
    }
}

// ld softmax + LH (bf16) + meanlh
__global__ void k_prep(const __bf16* __restrict__ ring_h, const float* __restrict__ ring_dis,
                       __bf16* __restrict__ LH, float* __restrict__ meanlh, int t) {
    int b = blockIdx.x;
    __shared__ float ld[CSZ];
    if (threadIdx.x == 0) {
        float vals[CSZ], cs = 0.f;
        for (int k = 0; k < CSZ; ++k) {
            int st = t - (CSZ - 1) + k;
            int slot = ((st % CSZ) + CSZ) % CSZ;
            cs += ring_dis[slot * BB + b];   // unwritten slots hold 0
            vals[k] = cs;
        }
        float mx = -1e30f;
        for (int k = 0; k < CSZ; ++k) mx = fmaxf(mx, vals[k]);
        float s = 0.f, e[CSZ];
        for (int k = 0; k < CSZ; ++k) { e[k] = __expf(vals[k] - mx); s += e[k]; }
        for (int k = 0; k < CSZ; ++k) ld[k] = e[k] / s;
    }
    __syncthreads();
    for (int cc = threadIdx.x; cc < HH; cc += blockDim.x) {
        float acc = 0.f;
        for (int k = 0; k < CSZ; ++k) {
            int st = t - (CSZ - 1) + k;
            int slot = ((st % CSZ) + CSZ) % CSZ;
            float hv = bf2f(ring_h[((size_t)slot * BB + b) * HH + cc]);
            float v = hv * ld[k];
            LH[(size_t)b * KH + k * HH + cc] = f2bf(v);
            acc += v;
        }
        meanlh[(size_t)b * HH + cc] = acc * (1.0f / (float)CSZ);
    }
}

__global__ void k_theme1(const float* __restrict__ meanlh, const float* __restrict__ sw,
                         const float* __restrict__ sb, float* __restrict__ tp) {
    int b = blockIdx.x, s = threadIdx.x;
    if (s >= HS) return;
    float acc = sb[s];
    const float* m = meanlh + (size_t)b * HH;
    for (int k = 0; k < HH; ++k) acc += m[k] * sw[(size_t)k * HS + s];
    tp[(size_t)b * HS + s] = acc;
}
__global__ void k_theme2(const float* __restrict__ tp, const float* __restrict__ rw,
                         const float* __restrict__ rb, float* __restrict__ theme) {
    int b = blockIdx.x, o = threadIdx.x;
    float acc = rb[o];
    const float* t = tp + (size_t)b * HS;
    for (int s = 0; s < HS; ++s) {
        float v = t[s]; v = v > 0.f ? v : 0.f;
        acc += v * rw[(size_t)s * HH + o];
    }
    theme[(size_t)b * HH + o] = sigmoidf(acc);
}

__global__ void k_finalize(const float* __restrict__ theme, const float* __restrict__ conv_out,
                           const __bf16* __restrict__ hbf, __bf16* __restrict__ rnnbf) {
    int i = blockIdx.x * blockDim.x + threadIdx.x;
    if (i >= BB * HH) return;
    float v = theme[i] * conv_out[i] + bf2f(hbf[i]);
    rnnbf[i] = f2bf(v);
}

// ---------------- launcher ----------------
extern "C" void kernel_launch(void* const* d_in, const int* in_sizes, int n_in,
                              void* d_out, int out_size, void* d_ws, size_t ws_size,
                              hipStream_t stream) {
    (void)in_sizes; (void)n_in; (void)out_size; (void)ws_size;
    const float* x        = (const float*)d_in[0];
    const float* timep    = (const float*)d_in[1];
    const float* kernel_w = (const float*)d_in[2];
    const float* kernel_b = (const float*)d_in[3];
    const float* rec_w    = (const float*)d_in[4];
    const float* rec_b    = (const float*)d_in[5];
    const float* scale_w  = (const float*)d_in[6];
    const float* scale_b  = (const float*)d_in[7];
    const float* rescale_w= (const float*)d_in[8];
    const float* rescale_b= (const float*)d_in[9];
    const float* conv_w   = (const float*)d_in[10];
    const float* conv_b   = (const float*)d_in[11];
    const float* out_w    = (const float*)d_in[12];
    const float* out_b    = (const float*)d_in[13];
    float* dout = (float*)d_out;

    size_t off = 0;
    auto alloc = [&](size_t bytes) -> char* {
        char* p = (char*)d_ws + off;
        off = (off + bytes + 255) & ~(size_t)255;
        return p;
    };
    __bf16* x_bf     = (__bf16*)alloc((size_t)BB * TT * INF_ * 2);
    __bf16* wcomb_t  = (__bf16*)alloc((size_t)GP * KC * 2);    // N-padded
    float*  bias_g   = (float*) alloc((size_t)GG * 4);
    __bf16* wflat_t  = (__bf16*)alloc((size_t)HH * KH * 2);
    __bf16* owt      = (__bf16*)alloc((size_t)OUTD * HH * 2);
    __bf16* Amat     = (__bf16*)alloc((size_t)BB * KC * 2);
    float*  xo       = (float*) alloc((size_t)BB * GG * 4);
    float*  cstate   = (float*) alloc((size_t)BB * HH * 4);
    __bf16* hbf      = (__bf16*)alloc((size_t)BB * HH * 2);
    __bf16* ring_h   = (__bf16*)alloc((size_t)CSZ * BB * HH * 2);
    float*  ring_dis = (float*) alloc((size_t)CSZ * BB * 4);
    __bf16* LH       = (__bf16*)alloc((size_t)BB * KH * 2);
    float*  meanlh   = (float*) alloc((size_t)BB * HH * 4);
    float*  tp       = (float*) alloc((size_t)BB * HS * 4);
    float*  theme    = (float*) alloc((size_t)BB * HH * 4);
    float*  conv_out = (float*) alloc((size_t)BB * HH * 4);
    __bf16* rnnbf    = (__bf16*)alloc((size_t)BB * HH * 2);

    const int TB = 256;
    // one-time conversions / weight builds
    {
        size_t n = (size_t)BB * TT * INF_;
        k_cvt_bf16<<<dim3((unsigned)((n + TB - 1) / TB)), TB, 0, stream>>>(x, x_bf, n);
        size_t nw = (size_t)GP * KC;
        k_build_wcomb<<<dim3((unsigned)((nw + TB - 1) / TB)), TB, 0, stream>>>(kernel_w, rec_w, wcomb_t);
        k_bias_g<<<dim3((GG + TB - 1) / TB), TB, 0, stream>>>(kernel_b, rec_b, bias_g);
        size_t nf = (size_t)HH * KH;
        k_build_wflat<<<dim3((unsigned)((nf + TB - 1) / TB)), TB, 0, stream>>>(conv_w, wflat_t);
        k_build_owt<<<dim3((OUTD * HH + TB - 1) / TB), TB, 0, stream>>>(out_w, owt);
        // zero state + rings (bf16 buffers zeroed through float view; counts are even)
        size_t zc = (size_t)BB * HH;
        k_zero_f<<<dim3((unsigned)((zc + TB - 1) / TB)), TB, 0, stream>>>(cstate, zc);
        size_t zh = (size_t)BB * HH / 2;
        k_zero_f<<<dim3((unsigned)((zh + TB - 1) / TB)), TB, 0, stream>>>((float*)hbf, zh);
        size_t zr = (size_t)CSZ * BB * HH / 2;
        k_zero_f<<<dim3((unsigned)((zr + TB - 1) / TB)), TB, 0, stream>>>((float*)ring_h, zr);
        size_t zd = (size_t)CSZ * BB;
        k_zero_f<<<dim3((unsigned)((zd + TB - 1) / TB)), TB, 0, stream>>>(ring_dis, zd);
    }

    float* dist_out = dout + (size_t)BB * TT * OUTD;

    for (int t = 0; t < TT; ++t) {
        // activation row [x_t, time, h, time, pad]
        k_buildA<<<dim3((BB * KC + TB - 1) / TB), TB, 0, stream>>>(x_bf, timep, hbf, Amat, t);
        // xo = A @ Wcomb + (kernel_b + rec_b)
        wmma_gemm_bf16<KC, KC, 0><<<dim3(GP / 64, BB / 64), 128, 0, stream>>>(
            Amat, wcomb_t, xo, GG, bias_g, BB, GG);
        // gates -> c, h, ring, distance
        k_gates<<<dim3(BB), 256, 0, stream>>>(xo, cstate, hbf, ring_h, ring_dis, dist_out, t);
        // ld, LH, meanlh
        k_prep<<<dim3(BB), 256, 0, stream>>>(ring_h, ring_dis, LH, meanlh, t);
        // theme MLP
        k_theme1<<<dim3(BB), 192, 0, stream>>>(meanlh, scale_w, scale_b, tp);
        k_theme2<<<dim3(BB), HH, 0, stream>>>(tp, rescale_w, rescale_b, theme);
        // conv_out = LH @ Wflat + conv_b
        wmma_gemm_bf16<KH, KH, 0><<<dim3(HH / 64, BB / 64), 128, 0, stream>>>(
            LH, wflat_t, conv_out, HH, conv_b, BB, HH);
        // rnn_t = theme*conv_out + h
        k_finalize<<<dim3((BB * HH + TB - 1) / TB), TB, 0, stream>>>(theme, conv_out, hbf, rnnbf);
        // output[:, t, :] = sigmoid(rnn_t @ out_w + out_b)
        wmma_gemm_bf16<HH, HH, 1><<<dim3(1, BB / 64), 128, 0, stream>>>(
            rnnbf, owt, dout + (size_t)t * OUTD, TT * OUTD, out_b, BB, OUTD);
    }
}
